// MoEMlp_79989470921139
// MI455X (gfx1250) — matrix operand order repaired and checked by reading, not compiled, not run on hardware
//
#include <hip/hip_runtime.h>
#include <math.h>

// ---------------------------------------------------------------------------
// MoE (noisy top-k gating) + dense expert FFN for MI455X (gfx1250, wave32).
// Heavy GEMMs use v_wmma_f32_16x16x32_bf16 (fp32 accumulate); gating math
// stays fp32 scalar. FFN: 193 GFLOP, weights 189MB (fits 192MB L2) ->
// compute bound. The 128x768 x-panel lives resident in LDS (gfx1250: 320KB
// per workgroup), staged once per block; w1/w2 panels stream through LDS.
// ---------------------------------------------------------------------------

#define E_    10
#define K_    4
#define NTOK  2048
#define D_    768
#define H_    3072

// padded LDS row strides (bf16 elems): keep ds_load_b128 16B-aligned while
// spreading the 16 fragment rows across distinct bank groups (64 banks x 4B)
#define XLD   776     // 1552B = 388 dw -> row i at bank offset 4i
#define WLD   40      // 80B   = 20 dw  -> 20i mod 64 all-distinct for i<16
#define HLD   136     // 272B  = 68 dw  -> 4i pattern, distinct

typedef __attribute__((ext_vector_type(16))) __bf16        v16bf;
typedef __attribute__((ext_vector_type(8)))  float         v8f;
typedef __attribute__((ext_vector_type(4)))  unsigned int  u32x4;
typedef __attribute__((ext_vector_type(2)))  __bf16        v2bf;

// fp32 -> bf16 (RNE)
__device__ __forceinline__ unsigned short f2bf(float x) {
    __bf16 h = (__bf16)x;
    return __builtin_bit_cast(unsigned short, h);
}
// pack two fp32 into one dword of bf16 pair: exactly one v_cvt_pk_bf16_f32
__device__ __forceinline__ unsigned int pack2bf(float a, float b) {
#if __has_builtin(__builtin_amdgcn_cvt_pk_bf16_f32)
    v2bf p = __builtin_amdgcn_cvt_pk_bf16_f32(a, b);
    return __builtin_bit_cast(unsigned int, p);
#else
    return (unsigned int)f2bf(a) | ((unsigned int)f2bf(b) << 16);
#endif
}

// Build a 16xbf16 WMMA fragment from two 16-byte LDS chunks.
__device__ __forceinline__ v16bf load_frag(const unsigned short* p0,
                                           const unsigned short* p1) {
    union { v16bf v; u32x4 q[2]; } f;
    f.q[0] = *(const u32x4*)p0;
    f.q[1] = *(const u32x4*)p1;
    return f.v;
}

__device__ __forceinline__ float gelu_tanh(float v) {
    return 0.5f * v * (1.0f + tanhf(0.7978845608028654f * (v + 0.044715f * v * v * v)));
}

// ---------------------------------------------------------------------------
// Kernel 0: zero the per-expert accumulators (ws is poisoned by harness).
// ---------------------------------------------------------------------------
__global__ void moe_init_kernel(float* imp, float* load) {
    int t = threadIdx.x;
    if (t < E_) { imp[t] = 0.0f; load[t] = 0.0f; }
}

// ---------------------------------------------------------------------------
// Kernel 1: gating. One thread per token; w_gate/w_noise staged in LDS.
// ---------------------------------------------------------------------------
__global__ __launch_bounds__(256)
void moe_gate_kernel(const float* __restrict__ x,
                     const float* __restrict__ w_gate,
                     const float* __restrict__ w_noise,
                     const float* __restrict__ noise,
                     float* __restrict__ gates,
                     float* __restrict__ imp,
                     float* __restrict__ loadv) {
    __shared__ float sWg[D_ * E_];
    __shared__ float sWn[D_ * E_];
    const int t = threadIdx.x;
    for (int i = t; i < D_ * E_; i += 256) { sWg[i] = w_gate[i]; sWn[i] = w_noise[i]; }
    __syncthreads();

    const int n = blockIdx.x * 256 + t;   // grid = 8 blocks -> 2048 tokens
    float clean[E_], raw[E_];
#pragma unroll
    for (int e = 0; e < E_; ++e) { clean[e] = 0.0f; raw[e] = 0.0f; }

    const float* xp = x + (size_t)n * D_;
    for (int d = 0; d < D_; ++d) {
        const float xv = xp[d];
#pragma unroll
        for (int e = 0; e < E_; ++e) {
            clean[e] = fmaf(xv, sWg[d * E_ + e], clean[e]);
            raw[e]   = fmaf(xv, sWn[d * E_ + e], raw[e]);
        }
    }

    float sd[E_], nz[E_];
#pragma unroll
    for (int e = 0; e < E_; ++e) {
        const float r  = raw[e];
        const float sp = fmaxf(r, 0.0f) + log1pf(expf(-fabsf(r)));   // softplus
        sd[e] = sp + 0.01f;
        nz[e] = clean[e] + noise[(size_t)n * E_ + e] * sd[e];
    }

    // top-5 selection (E=10, tiny)
    float tmp[E_];
#pragma unroll
    for (int e = 0; e < E_; ++e) tmp[e] = nz[e];
    float tv[K_ + 1]; int ti[K_ + 1];
#pragma unroll
    for (int i = 0; i < K_ + 1; ++i) {
        float best = -3.0e38f; int bi = 0;
#pragma unroll
        for (int e = 0; e < E_; ++e)
            if (tmp[e] > best) { best = tmp[e]; bi = e; }
        tv[i] = best; ti[i] = bi; tmp[bi] = -3.0e38f;
    }

    // softmax over top-4 (tv[0] is the max, already sorted)
    float g4[K_]; float ssum = 0.0f;
#pragma unroll
    for (int i = 0; i < K_; ++i) { g4[i] = expf(tv[i] - tv[0]); ssum += g4[i]; }
    const float inv = 1.0f / ssum;

    float grow[E_];
#pragma unroll
    for (int e = 0; e < E_; ++e) grow[e] = 0.0f;
#pragma unroll
    for (int i = 0; i < K_; ++i) grow[ti[i]] = g4[i] * inv;

    const float thr_in = tv[K_], thr_out = tv[K_ - 1];
#pragma unroll
    for (int e = 0; e < E_; ++e) {
        gates[(size_t)n * E_ + e] = grow[e];
        atomicAdd(&imp[e], grow[e]);
        const float thr = (nz[e] > thr_in) ? thr_in : thr_out;
        const float p   = 0.5f * (1.0f + erff((clean[e] - thr) / sd[e] * 0.7071067811865476f));
        atomicAdd(&loadv[e], p);
    }
}

// ---------------------------------------------------------------------------
// Kernel 2: aux loss: 0.01*(cv2(importance)+cv2(load)), cv2 = var(ddof=1)/mean^2
// ---------------------------------------------------------------------------
__global__ void moe_loss_kernel(const float* __restrict__ imp,
                                const float* __restrict__ loadv,
                                float* __restrict__ out_loss) {
    if (threadIdx.x == 0 && blockIdx.x == 0) {
        float cv[2];
        for (int a = 0; a < 2; ++a) {
            const float* v = (a == 0) ? imp : loadv;
            float m = 0.0f;
            for (int e = 0; e < E_; ++e) m += v[e];
            m *= (1.0f / E_);
            float s = 0.0f;
            for (int e = 0; e < E_; ++e) { const float d = v[e] - m; s += d * d; }
            const float var = s / (E_ - 1);
            cv[a] = var / (m * m + 1e-10f);
        }
        *out_loss = 0.01f * (cv[0] + cv[1]);
    }
}

// ---------------------------------------------------------------------------
// Kernel 3: fused expert FFN. Block = 128 tokens x 128 out-dims, 8 waves
// arranged 4(M) x 2(N); each wave owns 2x4 wmma tiles (32x64).
// x-panel (128x768 bf16) + gates tile resident in LDS for the whole block.
// ---------------------------------------------------------------------------
__global__ __launch_bounds__(256, 1)
void moe_ffn_kernel(const float* __restrict__ x,
                    const float* __restrict__ w1,
                    const float* __restrict__ b1,
                    const float* __restrict__ w2,
                    const float* __restrict__ b2,
                    const float* __restrict__ gates,
                    float* __restrict__ out) {
    __shared__ alignas(16) unsigned short sX[128 * XLD];  // resident A [m][k]  198,656B
    __shared__ alignas(16) unsigned short sW[128 * WLD];  // B panel  [n][k]     10,240B
    __shared__ alignas(16) unsigned short sH[128 * HLD];  // h chunk  [m][k]     34,816B
    __shared__ float sG[E_ * 128];                        // gates [e][row]       5,120B

    const int t    = threadIdx.x;
    const int lane = t & 31;
    const int w    = t >> 5;       // wave 0..7
    const int wm   = w >> 1;       // 0..3  (M)
    const int wn   = w & 1;        // 0..1  (N)
    const int n0   = blockIdx.x * 128;   // token base
    const int d0   = blockIdx.y * 128;   // out-dim base

    const int lrow = lane & 15;
    const int lhi  = lane >> 4;          // 0/1 lane half
    const int akb  = lhi << 3;           // A: K sub-base (0 / 8)
    const int bkb  = lhi << 3;           // B: K sub-base (0 / 8)

    // staging coordinates
    const int xr  = t >> 1;              // x: row 0..127
    const int wkr = (t >> 4) << 1;       // w: k-row pair base 0,2,..,30
    const int wcb = (t & 15) << 3;       // w: col base 0..120

    // ---- stage resident x panel (fp32 -> bf16) and gates tile, once ----
    {
        const float* gp = x + (size_t)(n0 + xr) * D_;
        unsigned short* dp = &sX[xr * XLD];
        for (int c = (t & 1) * 16; c < D_; c += 32) {
            u32x4 q0 = { pack2bf(gp[c],      gp[c + 1]),  pack2bf(gp[c + 2],  gp[c + 3]),
                         pack2bf(gp[c + 4],  gp[c + 5]),  pack2bf(gp[c + 6],  gp[c + 7]) };
            u32x4 q1 = { pack2bf(gp[c + 8],  gp[c + 9]),  pack2bf(gp[c + 10], gp[c + 11]),
                         pack2bf(gp[c + 12], gp[c + 13]), pack2bf(gp[c + 14], gp[c + 15]) };
            *(u32x4*)&dp[c]     = q0;
            *(u32x4*)&dp[c + 8] = q1;
        }
        for (int i = t; i < E_ * 128; i += 256) {
            const int e = i >> 7, r = i & 127;
            sG[i] = gates[(size_t)(n0 + r) * E_ + e];
        }
    }

    v8f oacc[2][4] = {};                 // final out accumulators

    for (int e = 0; e < E_; ++e) {
        v8f yacc[2][4] = {};

        for (int hc = 0; hc < H_ / 128; ++hc) {       // 24 H-chunks
            const int h0 = hc * 128;
            v8f hacc[2][4] = {};

            // ---- GEMM1: h_chunk = x[128x768] @ w1_e[768 x 128] ----
            for (int k1 = 0; k1 < D_; k1 += 32) {
                __syncthreads();
                {   // stage w1 tile transposed: global (k=d, n=h) -> LDS [n][k]
                    const float* g0 = w1 + ((size_t)e * D_ + (k1 + wkr)) * H_ + h0 + wcb;
                    const float* g1 = g0 + H_;
                    __builtin_prefetch(g0 + 32 * H_, 0, 1);   // next K-panel
#pragma unroll
                    for (int i = 0; i < 8; ++i)
                        *(unsigned int*)&sW[(wcb + i) * WLD + wkr] = pack2bf(g0[i], g1[i]);
                }
                __syncthreads();

                v16bf bf[4];
#pragma unroll
                for (int tn = 0; tn < 4; ++tn) {
                    const int bcol = wn * 64 + tn * 16 + lrow;
                    bf[tn] = load_frag(&sW[bcol * WLD + bkb], &sW[bcol * WLD + bkb + 16]);
                }
#pragma unroll
                for (int tm = 0; tm < 2; ++tm) {
                    const int arow = wm * 32 + tm * 16 + lrow;
                    v16bf af = load_frag(&sX[arow * XLD + k1 + akb],
                                         &sX[arow * XLD + k1 + 16 + akb]);
#pragma unroll
                    for (int tn = 0; tn < 4; ++tn)
                        hacc[tm][tn] = __builtin_amdgcn_wmma_f32_16x16x32_bf16(
                            false, af, false, bf[tn], (short)0, hacc[tm][tn], false, false);
                }
            }

            // ---- +b1, gelu, write bf16 h chunk to LDS [m][k] ----
            __syncthreads();
#pragma unroll
            for (int tm = 0; tm < 2; ++tm)
#pragma unroll
                for (int tn = 0; tn < 4; ++tn) {
                    const int colc = wn * 64 + tn * 16 + lrow;     // 0..127
                    const float bb = b1[(size_t)e * H_ + h0 + colc];
#pragma unroll
                    for (int r = 0; r < 8; ++r) {
                        const int row = wm * 32 + tm * 16 + r + (lhi << 3);
                        sH[row * HLD + colc] = f2bf(gelu_tanh(hacc[tm][tn][r] + bb));
                    }
                }
            __syncthreads();

            // ---- GEMM2: y += h_chunk[128x128] @ w2_e[128 x 128] ----
            for (int k2 = 0; k2 < 128; k2 += 32) {
                __syncthreads();
                {   // stage w2 tile transposed: global (k=h, n=d) -> LDS [n][k]
                    const float* g0 = w2 + ((size_t)e * H_ + (h0 + k2 + wkr)) * D_ + d0 + wcb;
                    const float* g1 = g0 + D_;
#pragma unroll
                    for (int i = 0; i < 8; ++i)
                        *(unsigned int*)&sW[(wcb + i) * WLD + wkr] = pack2bf(g0[i], g1[i]);
                }
                __syncthreads();

                v16bf bf[4];
#pragma unroll
                for (int tn = 0; tn < 4; ++tn) {
                    const int bcol = wn * 64 + tn * 16 + lrow;
                    bf[tn] = load_frag(&sW[bcol * WLD + bkb], &sW[bcol * WLD + bkb + 16]);
                }
#pragma unroll
                for (int tm = 0; tm < 2; ++tm) {
                    const int arow = wm * 32 + tm * 16 + lrow;
                    v16bf af = load_frag(&sH[arow * HLD + k2 + akb],
                                         &sH[arow * HLD + k2 + 16 + akb]);
#pragma unroll
                    for (int tn = 0; tn < 4; ++tn)
                        yacc[tm][tn] = __builtin_amdgcn_wmma_f32_16x16x32_bf16(
                            false, af, false, bf[tn], (short)0, yacc[tm][tn], false, false);
                }
            }
        }

        // ---- out += gate[row,e] * (y + b2_e) ----
#pragma unroll
        for (int tm = 0; tm < 2; ++tm)
#pragma unroll
            for (int tn = 0; tn < 4; ++tn) {
                const int col = wn * 64 + tn * 16 + lrow;
                const float b2v = b2[(size_t)e * D_ + d0 + col];
#pragma unroll
                for (int r = 0; r < 8; ++r) {
                    const int row = wm * 32 + tm * 16 + r + (lhi << 3);
                    oacc[tm][tn][r] += sG[e * 128 + row] * (yacc[tm][tn][r] + b2v);
                }
            }
    }

    // ---- store ----
#pragma unroll
    for (int tm = 0; tm < 2; ++tm)
#pragma unroll
        for (int tn = 0; tn < 4; ++tn) {
            const int col = wn * 64 + tn * 16 + lrow;
#pragma unroll
            for (int r = 0; r < 8; ++r) {
                const int row = wm * 32 + tm * 16 + r + (lhi << 3);
                out[(size_t)(n0 + row) * D_ + d0 + col] = oacc[tm][tn][r];
            }
        }
}

// ---------------------------------------------------------------------------
extern "C" void kernel_launch(void* const* d_in, const int* in_sizes, int n_in,
                              void* d_out, int out_size, void* d_ws, size_t ws_size,
                              hipStream_t stream) {
    const float* x       = (const float*)d_in[0];
    const float* w_gate  = (const float*)d_in[1];
    const float* w_noise = (const float*)d_in[2];
    const float* w1      = (const float*)d_in[3];
    const float* b1      = (const float*)d_in[4];
    const float* w2      = (const float*)d_in[5];
    const float* b2      = (const float*)d_in[6];
    const float* noise   = (const float*)d_in[7];

    float* out      = (float*)d_out;
    float* loss_out = out + (size_t)NTOK * D_;    // tuple: (out, loss)

    float* gates = (float*)d_ws;                  // NTOK * E_
    float* imp   = gates + (size_t)NTOK * E_;     // E_
    float* loadv = imp + E_;                      // E_

    moe_init_kernel<<<1, 32, 0, stream>>>(imp, loadv);
    moe_gate_kernel<<<NTOK / 256, 256, 0, stream>>>(x, w_gate, w_noise, noise,
                                                    gates, imp, loadv);
    moe_loss_kernel<<<1, 32, 0, stream>>>(imp, loadv, loss_out);
    moe_ffn_kernel<<<dim3(NTOK / 128, D_ / 128), 256, 0, stream>>>(
        x, w1, b1, w2, b2, gates, out);
}
